// Store_Class_Token_Attn_Wrapper_3118146257252
// MI455X (gfx1250) — compile-verified, matrix operand order checked by reading, not run on hardware
//
#include <hip/hip_runtime.h>
#include <hip/hip_bf16.h>
#include <math.h>

typedef __attribute__((ext_vector_type(16))) _Float16 v16h;
typedef __attribute__((ext_vector_type(8)))  _Float16 v8h;
typedef __attribute__((ext_vector_type(8)))  float    v8f;
typedef __attribute__((ext_vector_type(4)))  int      v4i;

#define BB    16
#define NN    577
#define CC    1024
#define HH    16
#define DD    64
#define NPAD  608            // 19 * 32: attention K-dim tiles cleanly into 16x16x32 WMMA
#define MROWS (BB * NN)      // 9232 = 577*16, multiple of 16

#define KSTEP  32            // GEMM K-step (one WMMA depth)
#define LDSROW 40            // padded LDS row pitch (halves): 16-row b128 frag reads conflict-free
#define KT     (CC / KSTEP)  // 32 K iterations

// ---- CDNA5 async global->LDS path (guarded so compile never breaks) --------
#if defined(__gfx1250__) && __has_builtin(__builtin_amdgcn_global_load_async_to_lds_b128)
#define HAVE_ASYNC 1
#else
#define HAVE_ASYNC 0
#endif

#if HAVE_ASYNC
#if __has_builtin(__builtin_amdgcn_s_wait_asynccnt)
#define WAIT_ASYNC(n) __builtin_amdgcn_s_wait_asynccnt(n)
#else
#define WAIT_ASYNC(n) asm volatile("s_wait_asynccnt %0" :: "n"(n) : "memory")
#endif
#else
#define WAIT_ASYNC(n) ((void)0)
#endif

// Copy 16B global -> LDS. Async (ASYNCcnt-tracked DMA) when available.
__device__ __forceinline__ void cp16_g2l(const _Float16* g, _Float16* l) {
#if HAVE_ASYNC
  __builtin_amdgcn_global_load_async_to_lds_b128(
      (__attribute__((address_space(1))) v4i*)(_Float16*)g,
      (__attribute__((address_space(3))) v4i*)l, 0, 0);
#else
  *(v8h*)l = *(const v8h*)g;
#endif
}

// ---------------------------------------------------------------------------
// WMMA fragment loader for 16-bit A (16x32, rows K-contiguous) from a
// row-major matrix (global or LDS). Per ISA 7.12.2: lane L(<16) holds row M=L
// with K = {0..7, 16..23}; lane L+16 holds row M=L with K = {8..15, 24..31}.
// B-fragments for D = A x B with B = W^T (W row-major [N,K]) use the SAME
// pattern (B column n == W row n), so one loader serves both operands.
// ---------------------------------------------------------------------------
__device__ __forceinline__ v16h load_frag(const _Float16* __restrict__ p, int ld, int lane) {
  const int r  = lane & 15;
  const int kh = (lane >> 4) << 3;          // 0 or 8
  const _Float16* base = p + (size_t)r * ld + kh;
  v8h lo = *(const v8h*)(base);             // K = kh .. kh+7
  v8h hi = *(const v8h*)(base + 16);        // K = kh+16 .. kh+23
  v16h f;
#pragma unroll
  for (int i = 0; i < 8; ++i) { f[i] = lo[i]; f[i + 8] = hi[i]; }
  return f;
}

__device__ __forceinline__ v8f wmma_f16(v16h a, v16h b, v8f c) {
  return __builtin_amdgcn_wmma_f32_16x16x32_f16(false, a, false, b, (short)0, c, false, false);
}

// ---------------------------------------------------------------------------
// f32 -> f16 conversion (n must be a multiple of 4)
// ---------------------------------------------------------------------------
__global__ void cvt_f32_f16(const float* __restrict__ src, _Float16* __restrict__ dst, int n) {
  int idx = (blockIdx.x * blockDim.x + threadIdx.x) * 4;
  if (idx + 3 < n) {
    float4 v = *(const float4*)(src + idx);
    dst[idx + 0] = (_Float16)v.x; dst[idx + 1] = (_Float16)v.y;
    dst[idx + 2] = (_Float16)v.z; dst[idx + 3] = (_Float16)v.w;
  }
}

// ---------------------------------------------------------------------------
// Double-buffered LDS-staged GEMM core. Block = 256 threads = 8 waves laid
// out 2(M) x 4(J); block tile 32(M) x 256(J); K-step 32. Each 32x32 A tile
// and 256x32 B tile is staged into LDS ONCE per block (vs 4x/2x redundant
// global fragment fetches), with the tile-(k+1) async DMA overlapping the
// tile-k WMMAs. Every wave issues exactly 5 async ops per stage so the
// s_wait_asynccnt target is uniform.
// ---------------------------------------------------------------------------
__device__ __forceinline__ void stage_tiles(
    const _Float16* __restrict__ Ag, const _Float16* __restrict__ Bg,
    int kk, _Float16* Ab, _Float16* Bb, int tid, int mbase, int jbase) {
  // A tile: 32 rows x 32 halves = 128 x 16B chunks; tids 128..255 duplicate.
  {
    const int c  = tid & 127;
    const int ar = c >> 2;
    const int aq = (c & 3) * 8;
    int gr = mbase + ar;
    if (gr > MROWS - 1) gr = MROWS - 1;          // clamp OOB rows (last block)
    cp16_g2l(Ag + (size_t)gr * CC + kk + aq, Ab + ar * LDSROW + aq);
  }
  // B tile: 256 rows x 32 halves = 1024 x 16B chunks; 4 per thread.
#pragma unroll
  for (int q = 0; q < 4; ++q) {
    const int idx = tid + q * 256;
    const int br  = idx >> 2;
    const int bq  = (idx & 3) * 8;
    cp16_g2l(Bg + (size_t)(jbase + br) * CC + kk + bq, Bb + br * LDSROW + bq);
  }
}

__device__ __forceinline__ void gemm_core(
    const _Float16* __restrict__ Ag, const _Float16* __restrict__ Bg,
    _Float16* Abuf, _Float16* Bbuf, int mbase, int jbase, v8f acc[4]) {
  const int tid  = threadIdx.x;
  const int lane = tid & 31;
  const int wave = tid >> 5;
  const int wm = wave >> 2;      // 0..1
  const int wj = wave & 3;       // 0..3
  const int ASZ = 32 * LDSROW;
  const int BSZ = 256 * LDSROW;

  stage_tiles(Ag, Bg, 0, Abuf, Bbuf, tid, mbase, jbase);
  for (int kt = 0; kt < KT; ++kt) {
    const int cur = kt & 1;
    if (kt + 1 < KT)
      stage_tiles(Ag, Bg, (kt + 1) * KSTEP,
                  Abuf + (cur ^ 1) * ASZ, Bbuf + (cur ^ 1) * BSZ,
                  tid, mbase, jbase);
    if (kt + 1 < KT) { WAIT_ASYNC(5); } else { WAIT_ASYNC(0); }
    __syncthreads();                             // tile kt visible to all waves

    const _Float16* Ab = Abuf + cur * ASZ + (wm * 16) * LDSROW;
    const _Float16* Bb = Bbuf + cur * BSZ + (wj * 64) * LDSROW;
    v16h a = load_frag(Ab, LDSROW, lane);
#pragma unroll
    for (int t = 0; t < 4; ++t) {
      v16h b = load_frag(Bb + t * 16 * LDSROW, LDSROW, lane);
      acc[t] = wmma_f16(a, b, acc[t]);
    }
    __syncthreads();                             // done reading before overwrite
  }
}

// ---------------------------------------------------------------------------
// QKV GEMM: out[m, j] = x[m, :] . w_qkv[j, :] + b_qkv[j],  m<9232, j<3072.
// Scatters into q (pre-scaled 1/8), k [BH, NPAD, 64] and v^T [BH, 64, NPAD].
// ---------------------------------------------------------------------------
__global__ __launch_bounds__(256) void qkv_gemm(
    const _Float16* __restrict__ xh, const _Float16* __restrict__ wh,
    const float* __restrict__ bias,
    _Float16* __restrict__ qh, _Float16* __restrict__ kh, _Float16* __restrict__ vth) {
  __shared__ _Float16 Abuf[2 * 32 * LDSROW];
  __shared__ _Float16 Bbuf[2 * 256 * LDSROW];
  const int lane = threadIdx.x & 31;
  const int wave = threadIdx.x >> 5;
  const int mbase = blockIdx.y * 32;
  const int jbase = blockIdx.x * 256;
  const int m0 = mbase + (wave >> 2) * 16;
  const int j0 = jbase + (wave & 3) * 64;

  v8f acc[4] = {};
  gemm_core(xh, wh, Abuf, Bbuf, mbase, jbase, acc);
  if (m0 >= MROWS) return;                       // after all barriers — safe

  const int colb  = lane & 15;
  const int rhalf = (lane >> 4) << 3;
#pragma unroll
  for (int t = 0; t < 4; ++t) {
#pragma unroll
    for (int i = 0; i < 8; ++i) {
      const int m = m0 + i + rhalf;
      const int j = j0 + t * 16 + colb;
      const float v = acc[t][i] + bias[j];
      const int b = m / NN;
      const int n = m - b * NN;
      const int sel = j >> 10;
      const int jj  = j & 1023;
      const int h = jj >> 6;
      const int d = jj & 63;
      const size_t bh = (size_t)(b * HH + h);
      if (sel == 0)      qh[(bh * NPAD + n) * DD + d] = (_Float16)(v * 0.125f);
      else if (sel == 1) kh[(bh * NPAD + n) * DD + d] = (_Float16)v;
      else               vth[(bh * DD + d) * NPAD + n] = (_Float16)v;
    }
  }
}

// ---------------------------------------------------------------------------
// Flash attention: one wave per (bh, 16-row query tile). Online softmax over
// 19 column-pairs (32 keys each). P tile staged through LDS to re-fragment
// into WMMA A-layout for the PV matmul (V^T gives contiguous B fragments).
// ---------------------------------------------------------------------------
__global__ __launch_bounds__(32) void flash_attn(
    const _Float16* __restrict__ qh, const _Float16* __restrict__ kh,
    const _Float16* __restrict__ vth, _Float16* __restrict__ aoh) {
  __shared__ _Float16 pbuf[16 * 32];
  const int lane = threadIdx.x;
  const int row0 = blockIdx.x * 16;
  const int bh   = blockIdx.y;
  const _Float16* qbase = qh  + (size_t)bh * NPAD * DD;
  const _Float16* kbase = kh  + (size_t)bh * NPAD * DD;
  const _Float16* vbase = vth + (size_t)bh * DD * NPAD;

  const v16h q0 = load_frag(qbase + (size_t)row0 * DD +  0, DD, lane);
  const v16h q1 = load_frag(qbase + (size_t)row0 * DD + 32, DD, lane);

  v8f o[4] = {};
  float mrow[8], lrow[8];
#pragma unroll
  for (int i = 0; i < 8; ++i) { mrow[i] = -3.0e38f; lrow[i] = 0.0f; }

  const int colb = lane & 15;
  const int hi   = lane >> 4;

  for (int ct = 0; ct < NPAD / 32; ++ct) {
    const int c0 = ct * 32;
    v16h kb0 = load_frag(kbase + (size_t)(c0)      * DD +  0, DD, lane);
    v16h kb1 = load_frag(kbase + (size_t)(c0)      * DD + 32, DD, lane);
    v16h kb2 = load_frag(kbase + (size_t)(c0 + 16) * DD +  0, DD, lane);
    v16h kb3 = load_frag(kbase + (size_t)(c0 + 16) * DD + 32, DD, lane);
    v8f s0 = {}, s1 = {};
    s0 = wmma_f16(q0, kb0, s0); s0 = wmma_f16(q1, kb1, s0);
    s1 = wmma_f16(q0, kb2, s1); s1 = wmma_f16(q1, kb3, s1);

    const int col0 = c0 + colb;
    const int col1 = c0 + 16 + colb;
#pragma unroll
    for (int i = 0; i < 8; ++i) {
      float a0 = (col0 < NN) ? s0[i] : -3.0e38f;
      float a1 = (col1 < NN) ? s1[i] : -3.0e38f;
      float tm = fmaxf(a0, a1);
      tm = fmaxf(tm, __shfl_xor(tm, 1, 32));
      tm = fmaxf(tm, __shfl_xor(tm, 2, 32));
      tm = fmaxf(tm, __shfl_xor(tm, 4, 32));
      tm = fmaxf(tm, __shfl_xor(tm, 8, 32));
      const float nm = fmaxf(mrow[i], tm);
      const float sc = __expf(mrow[i] - nm);
      mrow[i] = nm;
      const float p0 = __expf(a0 - nm);
      const float p1 = __expf(a1 - nm);
      float rs = p0 + p1;
      rs += __shfl_xor(rs, 1, 32);
      rs += __shfl_xor(rs, 2, 32);
      rs += __shfl_xor(rs, 4, 32);
      rs += __shfl_xor(rs, 8, 32);
      lrow[i] = lrow[i] * sc + rs;
#pragma unroll
      for (int t = 0; t < 4; ++t) o[t][i] *= sc;
      const int prow = i + hi * 8;
      pbuf[prow * 32 + colb]      = (_Float16)p0;
      pbuf[prow * 32 + 16 + colb] = (_Float16)p1;
    }
    v16h pf = load_frag(pbuf, 32, lane);
#pragma unroll
    for (int t = 0; t < 4; ++t) {
      v16h vf = load_frag(vbase + (size_t)(t * 16) * NPAD + c0, NPAD, lane);
      o[t] = wmma_f16(pf, vf, o[t]);
    }
  }

  const int b = bh >> 4, h = bh & 15;
#pragma unroll
  for (int i = 0; i < 8; ++i) {
    const int r = row0 + i + hi * 8;
    if (r < NN) {
      const float inv = 1.0f / lrow[i];
      const size_t rowp = ((size_t)b * NN + r) * CC + h * DD;
#pragma unroll
      for (int t = 0; t < 4; ++t)
        aoh[rowp + t * 16 + colb] = (_Float16)(o[t][i] * inv);
    }
  }
}

// ---------------------------------------------------------------------------
// Projection GEMM: out[m, j] = ao[m, :] . w_proj[j, :] + b_proj[j] (f32 out)
// ---------------------------------------------------------------------------
__global__ __launch_bounds__(256) void proj_gemm(
    const _Float16* __restrict__ ah, const _Float16* __restrict__ wh,
    const float* __restrict__ bias, float* __restrict__ out) {
  __shared__ _Float16 Abuf[2 * 32 * LDSROW];
  __shared__ _Float16 Bbuf[2 * 256 * LDSROW];
  const int lane = threadIdx.x & 31;
  const int wave = threadIdx.x >> 5;
  const int mbase = blockIdx.y * 32;
  const int jbase = blockIdx.x * 256;
  const int m0 = mbase + (wave >> 2) * 16;
  const int j0 = jbase + (wave & 3) * 64;

  v8f acc[4] = {};
  gemm_core(ah, wh, Abuf, Bbuf, mbase, jbase, acc);
  if (m0 >= MROWS) return;

  const int colb  = lane & 15;
  const int rhalf = (lane >> 4) << 3;
#pragma unroll
  for (int t = 0; t < 4; ++t) {
#pragma unroll
    for (int i = 0; i < 8; ++i) {
      const int m = m0 + i + rhalf;
      const int j = j0 + t * 16 + colb;
      out[(size_t)m * CC + j] = acc[t][i] + bias[j];
    }
  }
}

// ---------------------------------------------------------------------------
// CLS-row logits + per-head softmax (exact, tiny FLOPs). One block per (b,h).
// ---------------------------------------------------------------------------
__global__ __launch_bounds__(128) void cls_head(
    const _Float16* __restrict__ qh, const _Float16* __restrict__ kh,
    float* __restrict__ logits, float* __restrict__ probs) {
  const int bh = blockIdx.x;
  const int tid = threadIdx.x;
  __shared__ float qs[DD];
  __shared__ float red[128];
  const _Float16* q0 = qh + (size_t)bh * NPAD * DD;   // query row 0 (pre-scaled)
  const _Float16* kb = kh + (size_t)bh * NPAD * DD;
  if (tid < DD) qs[tid] = (float)q0[tid];
  __syncthreads();

  float lv[5], lmax = -3.0e38f;
#pragma unroll
  for (int it = 0; it < 5; ++it) {
    const int n = tid + it * 128;
    float s = -3.0e38f;
    if (n < NN) {
      const _Float16* kr = kb + (size_t)n * DD;
      float acc = 0.0f;
#pragma unroll
      for (int d = 0; d < DD; ++d) acc += qs[d] * (float)kr[d];
      s = acc;
      logits[(size_t)bh * NN + n] = acc;
    }
    lv[it] = s;
    lmax = fmaxf(lmax, s);
  }
  red[tid] = lmax; __syncthreads();
  for (int off = 64; off > 0; off >>= 1) {
    if (tid < off) red[tid] = fmaxf(red[tid], red[tid + off]);
    __syncthreads();
  }
  const float M = red[0]; __syncthreads();
  float ev[5], ls = 0.0f;
#pragma unroll
  for (int it = 0; it < 5; ++it) {
    ev[it] = (lv[it] > -1.0e37f) ? __expf(lv[it] - M) : 0.0f;
    ls += ev[it];
  }
  red[tid] = ls; __syncthreads();
  for (int off = 64; off > 0; off >>= 1) {
    if (tid < off) red[tid] += red[tid + off];
    __syncthreads();
  }
  const float inv = 1.0f / red[0];
#pragma unroll
  for (int it = 0; it < 5; ++it) {
    const int n = tid + it * 128;
    if (n < NN) probs[(size_t)bh * NN + n] = ev[it] * inv;
  }
}

__global__ void cls_mean(const float* __restrict__ logits, const float* __restrict__ probs,
                         float* __restrict__ out_attn, float* __restrict__ out_logits) {
  const int b = blockIdx.x;
  const int n = blockIdx.y * 256 + threadIdx.x;
  if (n >= NN) return;
  float sl = 0.0f, sp = 0.0f;
#pragma unroll
  for (int h = 0; h < HH; ++h) {
    sl += logits[(size_t)(b * HH + h) * NN + n];
    sp += probs [(size_t)(b * HH + h) * NN + n];
  }
  out_logits[b * NN + n] = sl * (1.0f / 16.0f);
  out_attn  [b * NN + n] = sp * (1.0f / 16.0f);
}

// ---------------------------------------------------------------------------
extern "C" void kernel_launch(void* const* d_in, const int* in_sizes, int n_in,
                              void* d_out, int out_size, void* d_ws, size_t ws_size,
                              hipStream_t stream) {
  const float* x      = (const float*)d_in[0];
  const float* w_qkv  = (const float*)d_in[1];
  const float* b_qkv  = (const float*)d_in[2];
  const float* w_proj = (const float*)d_in[3];
  const float* b_proj = (const float*)d_in[4];

  float* out     = (float*)d_out;                       // [B, N, C]
  float* out_cta = out + (size_t)MROWS * CC;            // class_token_attention [B, N]
  float* out_ctl = out_cta + MROWS;                     // class_token_logits    [B, N]

  char* ws = (char*)d_ws;
  auto alloc = [&](size_t bytes) -> char* {
    char* p = ws;
    ws += (bytes + 255) & ~(size_t)255;
    return p;
  };
  _Float16* xh     = (_Float16*)alloc((size_t)MROWS * CC * 2);
  _Float16* wqkvh  = (_Float16*)alloc((size_t)3 * CC * CC * 2);
  _Float16* wprojh = (_Float16*)alloc((size_t)CC * CC * 2);
  _Float16* qh     = (_Float16*)alloc((size_t)BB * HH * NPAD * DD * 2);
  _Float16* khb    = (_Float16*)alloc((size_t)BB * HH * NPAD * DD * 2);
  _Float16* vth    = (_Float16*)alloc((size_t)BB * HH * DD * NPAD * 2);
  _Float16* aoh    = (_Float16*)alloc((size_t)MROWS * CC * 2);
  float*    clsl   = (float*)alloc((size_t)BB * HH * NN * 4);
  float*    clsp   = (float*)alloc((size_t)BB * HH * NN * 4);

  // 1) f32 -> f16 conversions
  {
    int n = MROWS * CC;
    cvt_f32_f16<<<(n / 4 + 255) / 256, 256, 0, stream>>>(x, xh, n);
    n = 3 * CC * CC;
    cvt_f32_f16<<<(n / 4 + 255) / 256, 256, 0, stream>>>(w_qkv, wqkvh, n);
    n = CC * CC;
    cvt_f32_f16<<<(n / 4 + 255) / 256, 256, 0, stream>>>(w_proj, wprojh, n);
  }
  // 2) QKV GEMM (WMMA, async-LDS staged) -> q (scaled), k, v^T
  qkv_gemm<<<dim3(3 * CC / 256, (MROWS + 31) / 32), 256, 0, stream>>>(
      xh, wqkvh, b_qkv, qh, khb, vth);
  // 3) flash attention (WMMA) -> attention output f16
  flash_attn<<<dim3((NN + 15) / 16, BB * HH), 32, 0, stream>>>(qh, khb, vth, aoh);
  // 4) CLS-row logits / softmax per head + head-mean
  cls_head<<<dim3(BB * HH), 128, 0, stream>>>(qh, khb, clsl, clsp);
  cls_mean<<<dim3(BB, 3), 256, 0, stream>>>(clsl, clsp, out_cta, out_ctl);
  // 5) projection GEMM (WMMA, async-LDS staged) + bias -> f32 output
  proj_gemm<<<dim3(CC / 256, (MROWS + 31) / 32), 256, 0, stream>>>(
      aoh, wprojh, b_proj, out);
}
